// DeepSpeechRNN_25091198943691
// MI455X (gfx1250) — matrix operand order repaired and checked by reading, not compiled, not run on hardware
//
#include <hip/hip_runtime.h>

// ---------------------------------------------------------------------------
// DeepSpeech bidirectional RNN for MI455X (gfx1250), persistent-kernel design.
//
//   inputs [B=32, T=800, F=161] f32,  W [161,1024] f32,  U [1024,1024] f32,
//   b [1024] f32.   out = hf + hb, each h = final state of an 800-step scan
//   h = min(relu(x_t @ W + b + h @ U), 20).
//
// Strategy:
//   * 32 persistent workgroups: blocks 0..15 = forward dir, 16..31 = backward.
//   * Each WG owns a 64-column slice of the 1024 hidden units. Its U-slice
//     (1024x64, bf16 = 128KB) and W-slice (192x64 padded, bf16 = 24KB) are
//     swizzled ONCE into LDS in exact WMMA-B fragment order (320KB WGP LDS).
//   * Per timestep: x_t arrives via the Tensor Data Mover (tensor_load_to_lds,
//     issued one step ahead into a double-buffered fp32 LDS stage, waited with
//     s_wait_tensorcnt), is converted LDS->LDS to bf16, then 8 waves
//     (2 M-tiles x 4 N-tiles) each run 32+6 v_wmma_f32_16x16x32_bf16 ops
//     reading A-fragments of h (bf16, global/L2) and x (LDS).
//   * Grid sync per step via per-step arrive counters (agent-scope
//     release/acquire atomics, s_sleep backoff). h ping-pongs in d_ws (bf16).
//   * Final tiny kernel sums the two direction-final states into f32 d_out.
// ---------------------------------------------------------------------------

typedef __attribute__((ext_vector_type(16))) __bf16 v16bf;
typedef __attribute__((ext_vector_type(8)))  __bf16 v8bf;
typedef __attribute__((ext_vector_type(8)))  float  v8f;
typedef __attribute__((ext_vector_type(4)))  unsigned int ui4;
typedef __attribute__((ext_vector_type(8)))  int i8v;
typedef __attribute__((ext_vector_type(4)))  int i4v;

namespace {
constexpr int Bb    = 32;    // batch
constexpr int Tt    = 800;   // timesteps
constexpr int Ff    = 161;   // features
constexpr int FPAD  = 192;   // features padded to 6 k-tiles of 32
constexpr int UD    = 1024;  // hidden units
constexpr int NWG   = 16;    // workgroups per direction
constexpr int COLS  = 64;    // hidden columns per WG
constexpr int NT    = 4;     // 16-wide N tiles per WG
constexpr int KTU   = UD / 32;    // 32 k-tiles over hidden dim
constexpr int KTW   = FPAD / 32;  // 6  k-tiles over feature dim
constexpr float CLIPV = 20.0f;

// LDS layout: Upre 131072B | Wpre 24576B | xin 12288B | stage 2x(32x161 f32)
constexpr int UPRE_ELEMS = KTU * NT * 32 * 16;   // 65536 bf16
constexpr int WPRE_ELEMS = KTW * NT * 32 * 16;   // 12288 bf16
constexpr int XIN_ELEMS  = Bb * FPAD;            // 6144  bf16
constexpr int STAGE_ELEMS = Bb * Ff;             // 5152  f32 per buffer
constexpr size_t LDS_BYTES =
    (UPRE_ELEMS + WPRE_ELEMS + XIN_ELEMS) * sizeof(__bf16)
    + 2 * STAGE_ELEMS * sizeof(float);           // 209152 bytes < 320KB WGP

// ws layout: [0,8192) step counters, then bf16 h buffers (2 dir x 2 pp x 32x1024)
constexpr size_t WS_CNT_BYTES = 8192;
constexpr size_t WS_H_ELEMS   = 2 * 2 * Bb * UD;
constexpr size_t WS_INIT_BYTES = WS_CNT_BYTES + WS_H_ELEMS * sizeof(__bf16);
}

union Frag { v16bf v; v8bf h[2]; };

// Swizzled B-fragment address: tile (kt,nt), element (kin = k%32, nl = n%16).
// Lane = nl + 16*(kin/16) holds 16 contiguous bf16 covering its k-run.
__device__ __forceinline__ int bswz(int kt, int nt, int kin, int nl) {
  int lane = nl + 16 * (kin >> 4);
  int i    = kin & 15;
  return (((kt * NT + nt) * 32) + lane) * 16 + i;
}

// Issue one TDM 2-D tile load: 32 rows x 161 f32, row stride 800*161 elems,
// from global `gptr` into LDS byte offset `lds_off`. Descriptor per ISA 8.3/8.4.
__device__ __forceinline__ void tdm_load_x(const float* gptr, unsigned lds_off) {
  unsigned long long ga = (unsigned long long)(uintptr_t)gptr;
  ui4 g0;
  g0[0] = 1u;                                            // count=1, user mode
  g0[1] = lds_off;                                       // lds_addr (bytes)
  g0[2] = (unsigned)(ga & 0xffffffffu);                  // global_addr[31:0]
  g0[3] = (unsigned)((ga >> 32) & 0x01ffffffu)           // global_addr[56:32]
          | (2u << 30);                                  // type = 2 ("image")
  i8v g1;
  g1[0] = (int)(2u << 16);            // workgroup_mask=0, data_size=2 (4B)
  g1[1] = (int)((unsigned)Ff << 16);  // tensor_dim0[15:0] = 161
  g1[2] = (int)((unsigned)Bb << 16);  // tensor_dim0 hi=0 | tensor_dim1 lo = 32
  g1[3] = (int)((unsigned)Ff << 16);  // tensor_dim1 hi=0 | tile_dim0 = 161
  g1[4] = Bb;                         // tile_dim1 = 32, tile_dim2 = 0
  g1[5] = Tt * Ff;                    // tensor_dim0_stride[31:0] = 128800
  g1[6] = 0;                          // stride0 hi | stride1 lo
  g1[7] = 0;
  i4v z4 = {0, 0, 0, 0};
  i8v z8 = {0, 0, 0, 0, 0, 0, 0, 0};
  // clang-23 / therock-10.0 6-arg form: (g0, g1, g2, g3, extra, cpol)
  __builtin_amdgcn_tensor_load_to_lds(g0, g1, z4, z4, z8, 0);
}

__global__ void __launch_bounds__(256, 1)
rnn_persistent(const float* __restrict__ inputs, const float* __restrict__ Wm,
               const float* __restrict__ Um, const float* __restrict__ bias,
               __bf16* __restrict__ hbuf, unsigned* __restrict__ cnt) {
  extern __shared__ char smem[];
  __bf16* Upre  = reinterpret_cast<__bf16*>(smem);
  __bf16* Wpre  = Upre + UPRE_ELEMS;
  __bf16* xin   = Wpre + WPRE_ELEMS;
  float*  stage = reinterpret_cast<float*>(xin + XIN_ELEMS);  // 2 buffers

  const int tid   = threadIdx.x;
  const int lane  = tid & 31;
  const int wave  = tid >> 5;       // 8 waves
  const int mt    = wave >> 2;      // 0..1  (M tile)
  const int nt    = wave & 3;       // 0..3  (N tile)
  const int dir   = blockIdx.x >> 4;        // 0 fwd, 1 bwd
  const int wgd   = blockIdx.x & 15;        // WG id within direction
  const int nbase = wgd * COLS;             // first hidden column owned

  // LDS byte offsets of the two x staging buffers (low 32 bits of generic ptr).
  const unsigned stage_off0 = (unsigned)(uintptr_t)(void*)stage;
  const unsigned stage_off1 = stage_off0 + STAGE_ELEMS * sizeof(float);

  // Kick TDM for the first timestep's input tile while we swizzle U/W.
  if (tid == 0) {
    int t0 = dir ? (Tt - 1) : 0;
    tdm_load_x(inputs + (size_t)t0 * Ff, stage_off0);
  }

  // ---- One-time: swizzle U and W column slices into LDS (bf16) ----
  for (int idx = tid; idx < UD * COLS; idx += 256) {
    int k = idx / COLS, n = idx % COLS;
    float v = Um[(size_t)k * UD + nbase + n];
    Upre[bswz(k >> 5, n >> 4, k & 31, n & 15)] = (__bf16)v;
  }
  for (int idx = tid; idx < FPAD * COLS; idx += 256) {
    int k = idx / COLS, n = idx % COLS;
    float v = (k < Ff) ? Wm[(size_t)k * UD + nbase + n] : 0.0f;
    Wpre[bswz(k >> 5, n >> 4, k & 31, n & 15)] = (__bf16)v;
  }
  __syncthreads();

  const int mrow = mt * 16 + (lane & 15);   // A-fragment row (batch index)
  const int cA0  = 16 * (lane >> 4);        // A k-offset: contiguous k = 16h + s
  const int nl   = lane & 15;
  const int col  = nbase + nt * 16 + nl;    // hidden column this lane emits
  const float bv = bias[col];

  for (int s = 0; s < Tt; ++s) {
    // ---- x_t staged by TDM: wait (issuer wave), publish, convert to bf16 ----
    if (tid == 0) __builtin_amdgcn_s_wait_tensorcnt((short)0);
    __syncthreads();
    const float* stg = stage + (s & 1) * STAGE_ELEMS;
    for (int idx = tid; idx < Bb * FPAD; idx += 256) {
      int bb = idx / FPAD, f = idx % FPAD;
      float v = (f < Ff) ? stg[bb * Ff + f] : 0.0f;
      xin[bb * FPAD + f] = (__bf16)v;
    }
    __syncthreads();   // xin ready; stage[(s+1)&1] reads from step s-1 retired
    // ---- Issue TDM for x_{t+1} into the other stage buffer (overlaps WMMA) --
    if (tid == 0 && s + 1 < Tt) {
      int t2 = dir ? (Tt - 2 - s) : (s + 1);
      tdm_load_x(inputs + (size_t)t2 * Ff, ((s + 1) & 1) ? stage_off1 : stage_off0);
    }

    const __bf16* hprev = hbuf + ((size_t)(dir * 2 + (s & 1))) * Bb * UD;
    __bf16*       hnext = hbuf + ((size_t)(dir * 2 + ((s + 1) & 1))) * Bb * UD;

    v8f acc = {};
    // ---- h @ U[:, slice]  (A from global bf16 h, B from LDS swizzle) ----
    const __bf16* hrow = hprev + (size_t)mrow * UD;
#pragma unroll 4
    for (int kt = 0; kt < KTU; ++kt) {
      Frag a, b;
      const __bf16* ab = hrow + kt * 32 + cA0;
      a.h[0] = *(const v8bf*)(ab);
      a.h[1] = *(const v8bf*)(ab + 8);
      const __bf16* bb2 = Upre + ((kt * NT + nt) * 32 + lane) * 16;
      b.h[0] = *(const v8bf*)(bb2);
      b.h[1] = *(const v8bf*)(bb2 + 8);
      acc = __builtin_amdgcn_wmma_f32_16x16x32_bf16(
          false, a.v, false, b.v, (short)0, acc, false, false);
    }
    // ---- x_t @ W[:, slice]  (A from LDS bf16 xin) ----
    const __bf16* xrow = xin + (size_t)mrow * FPAD;
#pragma unroll
    for (int kt = 0; kt < KTW; ++kt) {
      Frag a, b;
      const __bf16* ab = xrow + kt * 32 + cA0;
      a.h[0] = *(const v8bf*)(ab);
      a.h[1] = *(const v8bf*)(ab + 8);
      const __bf16* bb2 = Wpre + ((kt * NT + nt) * 32 + lane) * 16;
      b.h[0] = *(const v8bf*)(bb2);
      b.h[1] = *(const v8bf*)(bb2 + 8);
      acc = __builtin_amdgcn_wmma_f32_16x16x32_bf16(
          false, a.v, false, b.v, (short)0, acc, false, false);
    }

    // ---- Epilogue: clipped ReLU, write bf16 h_next slice ----
#pragma unroll
    for (int r = 0; r < 8; ++r) {
      int m = mt * 16 + r + ((lane >> 4) << 3);   // C/D layout: M = r + 8*(lane/16)
      float v = acc[r] + bv;
      v = v < 0.0f ? 0.0f : (v > CLIPV ? CLIPV : v);
      hnext[(size_t)m * UD + col] = (__bf16)v;
    }

    // ---- Grid-wide step barrier (per-direction, per-step counter) ----
    __threadfence();              // make this thread's h_next stores agent-visible
    __syncthreads();              // whole block flushed
    if (tid == 0) {
      unsigned* c = cnt + dir * Tt + s;
      __hip_atomic_fetch_add(c, 1u, __ATOMIC_RELEASE, __HIP_MEMORY_SCOPE_AGENT);
      while (__hip_atomic_load(c, __ATOMIC_ACQUIRE, __HIP_MEMORY_SCOPE_AGENT) < NWG)
        __builtin_amdgcn_s_sleep(2);
    }
    __syncthreads();
    __threadfence();              // acquire: don't read stale h_prev next step
  }
}

__global__ void __launch_bounds__(256)
combine_dirs(const __bf16* __restrict__ hbuf, float* __restrict__ out) {
  int i = blockIdx.x * 256 + threadIdx.x;
  if (i < Bb * UD) {
    // After 800 steps (even), final state lives in ping-pong slot 0 per dir.
    float hf = (float)hbuf[i];                              // dir0, pp0
    float hb = (float)hbuf[(size_t)2 * Bb * UD + i];        // dir1, pp0
    out[i] = hf + hb;
  }
}

extern "C" void kernel_launch(void* const* d_in, const int* in_sizes, int n_in,
                              void* d_out, int out_size, void* d_ws, size_t ws_size,
                              hipStream_t stream) {
  const float* inputs = (const float*)d_in[0];
  const float* Wm     = (const float*)d_in[1];
  const float* Um     = (const float*)d_in[2];
  const float* bias   = (const float*)d_in[3];
  float* out          = (float*)d_out;

  unsigned* cnt = (unsigned*)d_ws;
  __bf16*   hbuf = (__bf16*)((char*)d_ws + WS_CNT_BYTES);

  // Zero step counters + h ping-pong buffers (h0 = 0). Capture-legal.
  (void)hipMemsetAsync(d_ws, 0, WS_INIT_BYTES, stream);

  rnn_persistent<<<32, 256, LDS_BYTES, stream>>>(inputs, Wm, Um, bias, hbuf, cnt);
  combine_dirs<<<(Bb * UD + 255) / 256, 256, 0, stream>>>(hbuf, out);
  (void)in_sizes; (void)n_in; (void)out_size; (void)ws_size;
}
